// Eq2to2_44281112822426
// MI455X (gfx1250) — matrix operand order repaired and checked by reading, not compiled, not run on hardware
//
#include <hip/hip_runtime.h>
#include <hip/hip_bf16.h>

// Problem constants (from reference)
#define NB   8      // batch N
#define MM   128    // m
#define DD   64     // IN_DIM
#define SS   64     // OUT_DIM
#define CB   15     // BASIS (coefs stride: d: SS*CB=960, s: CB)

typedef __attribute__((ext_vector_type(16))) __bf16 v16bf;
typedef __attribute__((ext_vector_type(8)))  float  v8f;

// ---------------------------------------------------------------------------
// Kernel 1: per (n,i) row-sums, col-sums, diagonal over d
// grid: NB*MM blocks, 64 threads (thread = d)
// ---------------------------------------------------------------------------
__global__ __launch_bounds__(64) void k_reduce(const float* __restrict__ x,
                                               float* __restrict__ rowsum,
                                               float* __restrict__ colsum,
                                               float* __restrict__ diagv) {
    int ni = blockIdx.x;            // n*MM + i
    int n  = ni >> 7;
    int i  = ni & (MM - 1);
    int d  = threadIdx.x;
    const float* base = x + (size_t)n * MM * MM * DD;
    float rs = 0.f, cs = 0.f;
    for (int j = 0; j < MM; ++j) {
        rs += base[((size_t)i * MM + j) * DD + d];
        cs += base[((size_t)j * MM + i) * DD + d];
    }
    rowsum[(size_t)ni * DD + d] = rs;
    colsum[(size_t)ni * DD + d] = cs;
    diagv [(size_t)ni * DD + d] = base[((size_t)i * MM + i) * DD + d];
}

// ---------------------------------------------------------------------------
// Kernel 2: per-n sum(diag) and sum(all) over d
// grid: NB blocks, 64 threads (thread = d)
// ---------------------------------------------------------------------------
__global__ __launch_bounds__(64) void k_persum(const float* __restrict__ rowsum,
                                               const float* __restrict__ diagv,
                                               float* __restrict__ sumdiag,
                                               float* __restrict__ sumall) {
    int n = blockIdx.x;
    int d = threadIdx.x;
    float sd = 0.f, sa = 0.f;
    for (int i = 0; i < MM; ++i) {
        sd += diagv [((size_t)n * MM + i) * DD + d];
        sa += rowsum[((size_t)n * MM + i) * DD + d];
    }
    sumdiag[(size_t)n * DD + d] = sd;
    sumall [(size_t)n * DD + d] = sa;
}

// ---------------------------------------------------------------------------
// Kernel 3: fold 13 non-GEMM basis ops (+bias/diag_bias) into R, C, Dg
//   R [n,i,s] = diag_i·W4 + colsum_i·W11 + rowsum_i·W12
//   C [n,j,s] = diag_j·W3 + colsum_j·W9  + rowsum_j·W10 + Σdiag·W13 + Σall·W14 + bias
//   Dg[n,i,s] = diag_i·W2 + rowsum_i·W6  + colsum_i·W7  + Σdiag·W5  + Σall·W8  + diag_bias
// grid: NB*MM blocks, 64 threads (thread = s)
// ---------------------------------------------------------------------------
__global__ __launch_bounds__(64) void k_rcd(const float* __restrict__ coefs,
                                            const float* __restrict__ rowsum,
                                            const float* __restrict__ colsum,
                                            const float* __restrict__ diagv,
                                            const float* __restrict__ sumdiag,
                                            const float* __restrict__ sumall,
                                            const float* __restrict__ bias,
                                            const float* __restrict__ diag_bias,
                                            float* __restrict__ R,
                                            float* __restrict__ C,
                                            float* __restrict__ Dg) {
    int ni = blockIdx.x;
    int n  = ni >> 7;
    int s  = threadIdx.x;
    float r = 0.f, c = 0.f, dg = 0.f;
    for (int d = 0; d < DD; ++d) {
        const float* w = coefs + (size_t)d * (SS * CB) + (size_t)s * CB;
        float di = diagv  [(size_t)ni * DD + d];
        float rs = rowsum [(size_t)ni * DD + d];
        float cs = colsum [(size_t)ni * DD + d];
        float sd = sumdiag[(size_t)n  * DD + d];
        float sa = sumall [(size_t)n  * DD + d];
        r  = fmaf(di, w[4],  fmaf(cs, w[11], fmaf(rs, w[12], r)));
        c  = fmaf(di, w[3],  fmaf(cs, w[9],  fmaf(rs, w[10],
             fmaf(sd, w[13], fmaf(sa, w[14], c)))));
        dg = fmaf(di, w[2],  fmaf(rs, w[6],  fmaf(cs, w[7],
             fmaf(sd, w[5],  fmaf(sa, w[8],  dg)))));
    }
    R [(size_t)ni * SS + s] = r;
    C [(size_t)ni * SS + s] = c  + bias[s];
    Dg[(size_t)ni * SS + s] = dg + diag_bias[s];
}

// ---------------------------------------------------------------------------
// Kernel 4: pack W0 (identity) and W1 (transpose) into WMMA B-fragment order.
// B matrix is 32x16 (KxN) bf16, v16bf per lane (ISA 7.12 B layout):
//   N = lane%16, K = 8*(lane/16) + e (e<8), 16 + 8*(lane/16) + (e-8) (e>=8)  (within one K=32 step; K-step ks adds 32)
// Wpack index: [((g*2 + ks)*4 + stile)*512 + lane*16 + e]
// grid: 1 block, 256 threads
// ---------------------------------------------------------------------------
__global__ __launch_bounds__(256) void k_pack(const float* __restrict__ coefs,
                                              __bf16* __restrict__ wpack) {
    for (int idx = threadIdx.x; idx < 512; idx += 256) {
        int lane  = idx & 31;
        int t     = idx >> 5;
        int stile = t & 3;  t >>= 2;
        int ks    = t & 1;  t >>= 1;
        int g     = t;                       // 0 -> W0 (b=0), 1 -> W1 (b=1)
        int s     = stile * 16 + (lane & 15);
        int kbase = ks * 32 + 8 * (lane >> 4);
        __bf16* dst = wpack + ((size_t)idx) * 16;
        #pragma unroll
        for (int e = 0; e < 16; ++e) {
            int k = kbase + (e < 8 ? e : e + 8);   // k in [0,64)
            dst[e] = (__bf16)coefs[(size_t)k * (SS * CB) + (size_t)s * CB + g];
        }
    }
}

// ---------------------------------------------------------------------------
// Kernel 5: main GEMM + fused epilogue.
// One wave per 16(j) x 64(s) tile of out[n,i,:,:]. 16 WMMAs per wave:
//   2 (A matrices) x 2 (K steps of 32) x 4 (s tiles of 16).
// A fragment (16x32 bf16, ISA 7.12 A layout): lane row m=lane%16,
//   K = ks*32 + 8*(lane>=16) + {e | e<8} and +16 more for e>=8.
// Native (__bf16) casts -> packed v_cvt bf16 conversions.
// ---------------------------------------------------------------------------
static __device__ __forceinline__ v16bf load_a_frag(const float* __restrict__ rowbase,
                                                    int koff) {
    // koff = ks*32 + 8*(lane>=16); picks 8 floats at koff and 8 at koff+16
    v16bf v;
    #pragma unroll
    for (int e = 0; e < 8; ++e)  v[e]     = (__bf16)rowbase[koff + e];
    #pragma unroll
    for (int e = 0; e < 8; ++e)  v[8 + e] = (__bf16)rowbase[koff + 16 + e];
    return v;
}

__global__ __launch_bounds__(256) void k_main(const float* __restrict__ x,
                                              const float* __restrict__ mask,
                                              const __bf16* __restrict__ wpack,
                                              const float* __restrict__ R,
                                              const float* __restrict__ C,
                                              const float* __restrict__ Dg,
                                              float* __restrict__ out) {
    int lane = threadIdx.x & 31;
    int wid  = blockIdx.x * (blockDim.x >> 5) + (threadIdx.x >> 5);   // [0, 8192)
    int jt   = wid & 7;
    int i    = (wid >> 3) & (MM - 1);
    int n    = wid >> 10;
    int j0   = jt * 16;
    int m    = lane & 15;     // A row within tile
    int kh   = lane >> 4;     // lane half

    const float* arow0 = x + (((size_t)(n * MM + i)) * MM + (j0 + m)) * DD;  // x[n,i,j0+m,:]
    const float* arow1 = x + (((size_t)(n * MM + j0 + m)) * MM + i) * DD;    // x[n,j0+m,i,:]

    v8f acc0 = {}, acc1 = {}, acc2 = {}, acc3 = {};

    #pragma unroll
    for (int g = 0; g < 2; ++g) {
        const float* arow = (g == 0) ? arow0 : arow1;
        #pragma unroll
        for (int ks = 0; ks < 2; ++ks) {
            v16bf a = load_a_frag(arow, ks * 32 + 8 * kh);
            const __bf16* wb = wpack + (size_t)((g * 2 + ks) * 4) * 512 + (size_t)lane * 16;
            v16bf b0 = *(const v16bf*)(wb);
            v16bf b1 = *(const v16bf*)(wb + 512);
            v16bf b2 = *(const v16bf*)(wb + 1024);
            v16bf b3 = *(const v16bf*)(wb + 1536);
            acc0 = __builtin_amdgcn_wmma_f32_16x16x32_bf16(false, a, false, b0,
                                                           (short)0, acc0, false, false);
            acc1 = __builtin_amdgcn_wmma_f32_16x16x32_bf16(false, a, false, b1,
                                                           (short)0, acc1, false, false);
            acc2 = __builtin_amdgcn_wmma_f32_16x16x32_bf16(false, a, false, b2,
                                                           (short)0, acc2, false, false);
            acc3 = __builtin_amdgcn_wmma_f32_16x16x32_bf16(false, a, false, b3,
                                                           (short)0, acc3, false, false);
        }
    }

    // Epilogue: D element p of lane holds output row M = p + 8*kh (j = j0 + ...),
    // column N = lane%16 (s = st*16 + ...).
    int scol = lane & 15;
    size_t nio = ((size_t)n * MM + i);
    size_t outbase = nio * MM * SS;
    float mk[8];
    #pragma unroll
    for (int p = 0; p < 8; ++p)
        mk[p] = mask[nio * MM + (j0 + p + 8 * kh)];

    v8f accs[4] = {acc0, acc1, acc2, acc3};
    #pragma unroll
    for (int st = 0; st < 4; ++st) {
        int s = st * 16 + scol;
        float rv = R[nio * SS + s];
        float dv = Dg[nio * SS + s];
        #pragma unroll
        for (int p = 0; p < 8; ++p) {
            int j = j0 + p + 8 * kh;
            float v = accs[st][p] + rv + C[((size_t)n * MM + j) * SS + s];
            if (i == j) v += dv;
            v = (v > 0.f) ? v : 0.01f * v;      // leaky_relu
            v *= mk[p];                          // mask
            out[outbase + (size_t)j * SS + s] = v;
        }
    }
}

// ---------------------------------------------------------------------------
// Launcher. Inputs (setup_inputs order):
//   d_in[0]=inputs f32 [N,M,M,64], d_in[1]=nobj i32 (unused),
//   d_in[2]=mask f32 [N,M,M,1], d_in[3]=coefs f32 [64,64,15],
//   d_in[4]=bias f32 [64], d_in[5]=diag_bias f32 [64]
// ---------------------------------------------------------------------------
extern "C" void kernel_launch(void* const* d_in, const int* in_sizes, int n_in,
                              void* d_out, int out_size, void* d_ws, size_t ws_size,
                              hipStream_t stream) {
    const float* x     = (const float*)d_in[0];
    const float* mask  = (const float*)d_in[2];
    const float* coefs = (const float*)d_in[3];
    const float* bias  = (const float*)d_in[4];
    const float* dbias = (const float*)d_in[5];
    float* outp = (float*)d_out;

    float* ws = (float*)d_ws;
    const size_t NM = (size_t)NB * MM;           // 1024
    float* rowsum  = ws;                          // NM*64
    float* colsum  = rowsum  + NM * DD;           // NM*64
    float* diagv   = colsum  + NM * DD;           // NM*64
    float* sumdiag = diagv   + NM * DD;           // NB*64
    float* sumall  = sumdiag + (size_t)NB * DD;   // NB*64
    float* R       = sumall  + (size_t)NB * DD;   // NM*64
    float* C       = R       + NM * SS;           // NM*64
    float* Dg      = C       + NM * SS;           // NM*64
    __bf16* wpack  = (__bf16*)(Dg + NM * SS);     // 8192 bf16

    k_reduce<<<NB * MM, 64, 0, stream>>>(x, rowsum, colsum, diagv);
    k_persum<<<NB, 64, 0, stream>>>(rowsum, diagv, sumdiag, sumall);
    k_rcd<<<NB * MM, 64, 0, stream>>>(coefs, rowsum, colsum, diagv, sumdiag, sumall,
                                      bias, dbias, R, C, Dg);
    k_pack<<<1, 256, 0, stream>>>(coefs, wpack);
    // 8192 wave-tiles / 8 waves per block = 1024 blocks
    k_main<<<1024, 256, 0, stream>>>(x, mask, wpack, R, C, Dg, outp);
}
